// Attention_58712202936937
// MI455X (gfx1250) — matrix operand order repaired and checked by reading, not compiled, not run on hardware
//
#include <hip/hip_runtime.h>
#include <hip/hip_bf16.h>

#define B_ 8
#define N_ 1024
#define H_ 12
#define D_ 64
#define C_ 768
#define QKV_ 2304
#define SCALE_ 0.125f

typedef __bf16 bf16;
typedef __attribute__((ext_vector_type(16))) __bf16 v16bf;
typedef __attribute__((ext_vector_type(8)))  float   v8f;

// D = A(16x32 bf16) * B(32x16 bf16) + C(16x16 f32)
#define WMMA_BF16(a, b, c) \
  __builtin_amdgcn_wmma_f32_16x16x32_bf16(false, (a), false, (b), (short)0, (c), false, false)

// reassemble a fragment from two raw 16B register chunks (no memory op)
static __device__ inline v16bf pack2(uint4 q0, uint4 q1) {
  union { v16bf v; uint4 q[2]; } u;
  u.q[0] = q0; u.q[1] = q1;
  return u.v;
}

// ============ Kernel 0: f32 -> bf16 bulk convert (8 elems/thread) ============
__global__ __launch_bounds__(256) void cvt_bf16_kernel(
    const float* __restrict__ src, bf16* __restrict__ dst, int n8) {
  int i = blockIdx.x * 256 + threadIdx.x;
  if (i >= n8) return;
  const float4* s = (const float4*)src + (size_t)i * 2;
  float4 f0 = s[0], f1 = s[1];
  union { uint4 q; bf16 e[8]; } u;
  u.e[0] = (bf16)f0.x; u.e[1] = (bf16)f0.y; u.e[2] = (bf16)f0.z; u.e[3] = (bf16)f0.w;
  u.e[4] = (bf16)f1.x; u.e[5] = (bf16)f1.y; u.e[6] = (bf16)f1.z; u.e[7] = (bf16)f1.w;
  ((uint4*)dst)[i] = u.q;
}

// ======== Kernel 1: QKV = x @ w_qkv^T + bias -> q (scaled) / k / v^T bf16 ========
// Wave tile: 32 (M) x 64 (N); double-buffered register pipeline over K.
__global__ __launch_bounds__(128) void qkv_gemm_kernel(
    const bf16* __restrict__ xbf, const bf16* __restrict__ wbf,
    const float* __restrict__ q_bias, const float* __restrict__ v_bias,
    bf16* __restrict__ qws, bf16* __restrict__ kws, bf16* __restrict__ vtws) {
  const int lane = threadIdx.x & 31;
  const int wave = threadIdx.x >> 5;
  const int lo = lane & 15, hi = lane >> 4;
  const int m0   = blockIdx.x * 32;
  const int col0 = blockIdx.y * 256 + wave * 64;

  v8f acc[2][4];
#pragma unroll
  for (int nt = 0; nt < 4; ++nt) {
    int c = col0 + nt * 16 + lo;
    int sel = c / C_;
    int hc = c - sel * C_;
    float bias = (sel == 0) ? q_bias[hc] : ((sel == 2) ? v_bias[hc] : 0.0f);
#pragma unroll
    for (int mt = 0; mt < 2; ++mt)
#pragma unroll
      for (int i = 0; i < 8; ++i) acc[mt][nt][i] = bias;
  }

  const bf16* ar0 = xbf + (size_t)(m0 + lo) * C_;
  const bf16* ar1 = xbf + (size_t)(m0 + 16 + lo) * C_;
  const bf16* br[4];
#pragma unroll
  for (int nt = 0; nt < 4; ++nt) br[nt] = wbf + (size_t)(col0 + nt * 16 + lo) * C_;

  uint4 ca[4], cb[8];
  // prime k0 = 0
  ca[0] = *(const uint4*)(ar0 + hi * 8);
  ca[1] = *(const uint4*)(ar0 + 16 + hi * 8);
  ca[2] = *(const uint4*)(ar1 + hi * 8);
  ca[3] = *(const uint4*)(ar1 + 16 + hi * 8);
#pragma unroll
  for (int nt = 0; nt < 4; ++nt) {
    cb[2*nt]   = *(const uint4*)(br[nt] + hi * 8);
    cb[2*nt+1] = *(const uint4*)(br[nt] + 16 + hi * 8);
  }

  for (int k0 = 0; k0 < C_; k0 += 32) {
    const int kn = (k0 + 32 < C_) ? (k0 + 32) : 0;  // branchless tail (uniform)
    uint4 na[4], nb[8];
    na[0] = *(const uint4*)(ar0 + kn + hi * 8);
    na[1] = *(const uint4*)(ar0 + kn + 16 + hi * 8);
    na[2] = *(const uint4*)(ar1 + kn + hi * 8);
    na[3] = *(const uint4*)(ar1 + kn + 16 + hi * 8);
#pragma unroll
    for (int nt = 0; nt < 4; ++nt) {
      nb[2*nt]   = *(const uint4*)(br[nt] + kn + hi * 8);
      nb[2*nt+1] = *(const uint4*)(br[nt] + kn + 16 + hi * 8);
    }

    v16bf a0 = pack2(ca[0], ca[1]);
    v16bf a1 = pack2(ca[2], ca[3]);
#pragma unroll
    for (int nt = 0; nt < 4; ++nt) {
      v16bf b = pack2(cb[2*nt], cb[2*nt+1]);
      acc[0][nt] = WMMA_BF16(a0, b, acc[0][nt]);
      acc[1][nt] = WMMA_BF16(a1, b, acc[1][nt]);
    }
#pragma unroll
    for (int i = 0; i < 4; ++i) ca[i] = na[i];
#pragma unroll
    for (int i = 0; i < 8; ++i) cb[i] = nb[i];
  }

#pragma unroll
  for (int nt = 0; nt < 4; ++nt) {
    int c = col0 + nt * 16 + lo;
    int sel = c / C_;
    int hc = c - sel * C_;
    int h = hc >> 6, d = hc & 63;
#pragma unroll
    for (int mt = 0; mt < 2; ++mt)
#pragma unroll
      for (int v = 0; v < 8; ++v) {
        int m = m0 + mt * 16 + v + 8 * hi;
        int b = m >> 10, nrow = m & (N_ - 1);
        float val = acc[mt][nt][v];
        if (sel == 0)
          qws[((size_t)(b * H_ + h) * N_ + nrow) * D_ + d] = (bf16)(val * SCALE_);
        else if (sel == 1)
          kws[((size_t)(b * H_ + h) * N_ + nrow) * D_ + d] = (bf16)val;
        else
          vtws[((size_t)(b * H_ + h) * D_ + d) * N_ + nrow] = (bf16)val;
      }
  }
}

// ================= Kernel 2: flash attention, one wave per 16-row q tile =================
__global__ __launch_bounds__(128) void flash_attn_kernel(
    const bf16* __restrict__ qws, const bf16* __restrict__ kws,
    const bf16* __restrict__ vtws, bf16* __restrict__ aout) {
  __shared__ bf16 plds[4 * 16 * 32];
  const int lane = threadIdx.x & 31;
  const int wave = threadIdx.x >> 5;
  const int lo = lane & 15, hi = lane >> 4;
  const int wid = blockIdx.x * 4 + wave;
  const int qt = wid & 63;
  const int bh = wid >> 6;
  const int b = bh / H_, h = bh - b * H_;

  const bf16* qp = qws  + (size_t)bh * N_ * D_;
  const bf16* kp = kws  + (size_t)bh * N_ * D_;
  const bf16* vp = vtws + (size_t)bh * D_ * N_;
  bf16* lds = plds + wave * 512;

  const bf16* qrow = qp + (size_t)(qt * 16 + lo) * D_;
  v16bf aq0 = pack2(*(const uint4*)(qrow + hi * 16),      *(const uint4*)(qrow + 8 + hi * 16));
  v16bf aq1 = pack2(*(const uint4*)(qrow + 32 + hi * 16), *(const uint4*)(qrow + 40 + hi * 16));

  float mrow[8], lrow[8];
  v8f o[4];
#pragma unroll
  for (int v = 0; v < 8; ++v) { mrow[v] = -1e30f; lrow[v] = 0.0f; }
#pragma unroll
  for (int nt = 0; nt < 4; ++nt)
#pragma unroll
    for (int v = 0; v < 8; ++v) o[nt][v] = 0.0f;

  // prime K fragments for key chunk 0
  v16bf bk[4];
  {
    const bf16* kr0 = kp + (size_t)lo * D_;
    const bf16* kr1 = kp + (size_t)(16 + lo) * D_;
    bk[0] = pack2(*(const uint4*)(kr0 + hi * 16),      *(const uint4*)(kr0 + 8 + hi * 16));
    bk[1] = pack2(*(const uint4*)(kr0 + 32 + hi * 16), *(const uint4*)(kr0 + 40 + hi * 16));
    bk[2] = pack2(*(const uint4*)(kr1 + hi * 16),      *(const uint4*)(kr1 + 8 + hi * 16));
    bk[3] = pack2(*(const uint4*)(kr1 + 32 + hi * 16), *(const uint4*)(kr1 + 40 + hi * 16));
  }

  for (int key0 = 0; key0 < N_; key0 += 32) {
    // S = Q @ K^T for the current 32-key chunk
    v8f s0, s1;
#pragma unroll
    for (int v = 0; v < 8; ++v) { s0[v] = 0.0f; s1[v] = 0.0f; }
    s0 = WMMA_BF16(aq0, bk[0], s0);
    s0 = WMMA_BF16(aq1, bk[1], s0);
    s1 = WMMA_BF16(aq0, bk[2], s1);
    s1 = WMMA_BF16(aq1, bk[3], s1);

    // issue V loads (this chunk) and K loads (next chunk) so they overlap softmax VALU
    uint4 vt[8], kn[8];
#pragma unroll
    for (int nt = 0; nt < 4; ++nt) {
      const bf16* vr = vp + (size_t)(nt * 16 + lo) * N_ + key0 + hi * 16;
      vt[2*nt]   = *(const uint4*)(vr);
      vt[2*nt+1] = *(const uint4*)(vr + 8);
    }
    {
      const int keyn = (key0 + 32 < N_) ? (key0 + 32) : 0;  // uniform, branchless
      const bf16* kr0 = kp + (size_t)(keyn + lo) * D_;
      const bf16* kr1 = kp + (size_t)(keyn + 16 + lo) * D_;
      kn[0] = *(const uint4*)(kr0 + hi * 16);      kn[1] = *(const uint4*)(kr0 + 8 + hi * 16);
      kn[2] = *(const uint4*)(kr0 + 32 + hi * 16); kn[3] = *(const uint4*)(kr0 + 40 + hi * 16);
      kn[4] = *(const uint4*)(kr1 + hi * 16);      kn[5] = *(const uint4*)(kr1 + 8 + hi * 16);
      kn[6] = *(const uint4*)(kr1 + 32 + hi * 16); kn[7] = *(const uint4*)(kr1 + 40 + hi * 16);
    }

    // online softmax; row (v + 8*hi) lives in the 16 lanes of this half-wave
#pragma unroll
    for (int v = 0; v < 8; ++v) {
      float mx = fmaxf(s0[v], s1[v]);
      mx = fmaxf(mx, __shfl_xor(mx, 1, 32));
      mx = fmaxf(mx, __shfl_xor(mx, 2, 32));
      mx = fmaxf(mx, __shfl_xor(mx, 4, 32));
      mx = fmaxf(mx, __shfl_xor(mx, 8, 32));
      float nm   = fmaxf(mrow[v], mx);
      float corr = __expf(mrow[v] - nm);
      float p0   = __expf(s0[v] - nm);
      float p1   = __expf(s1[v] - nm);
      float rs = p0 + p1;
      rs += __shfl_xor(rs, 1, 32);
      rs += __shfl_xor(rs, 2, 32);
      rs += __shfl_xor(rs, 4, 32);
      rs += __shfl_xor(rs, 8, 32);
      lrow[v] = lrow[v] * corr + rs;
      mrow[v] = nm;
      lds[(v + 8 * hi) * 32 + lo]      = (bf16)p0;
      lds[(v + 8 * hi) * 32 + 16 + lo] = (bf16)p1;
#pragma unroll
      for (int nt = 0; nt < 4; ++nt) o[nt][v] *= corr;
    }
    asm volatile("s_wait_dscnt 0" ::: "memory");   // P stores visible before A-frag reads

    v16bf ap = pack2(*(const uint4*)(lds + lo * 32 + hi * 16),
                     *(const uint4*)(lds + lo * 32 + 8 + hi * 16));
#pragma unroll
    for (int nt = 0; nt < 4; ++nt)
      o[nt] = WMMA_BF16(ap, pack2(vt[2*nt], vt[2*nt+1]), o[nt]);
    asm volatile("s_wait_dscnt 0" ::: "memory");   // WAR before next chunk's P store

    bk[0] = pack2(kn[0], kn[1]);
    bk[1] = pack2(kn[2], kn[3]);
    bk[2] = pack2(kn[4], kn[5]);
    bk[3] = pack2(kn[6], kn[7]);
  }

#pragma unroll
  for (int v = 0; v < 8; ++v) {
    float inv = 1.0f / lrow[v];
    int qr = qt * 16 + v + 8 * hi;
    bf16* orow = aout + ((size_t)b * N_ + qr) * C_ + h * D_;
#pragma unroll
    for (int nt = 0; nt < 4; ++nt)
      orow[nt * 16 + lo] = (bf16)(o[nt][v] * inv);
  }
}

// ======== Kernel 3: out = attn_out @ w_proj^T + b_proj (fp32 out), 32x64 wave tile ========
__global__ __launch_bounds__(128) void proj_gemm_kernel(
    const bf16* __restrict__ aout, const bf16* __restrict__ wbf,
    const float* __restrict__ b_proj, float* __restrict__ out) {
  const int lane = threadIdx.x & 31;
  const int wave = threadIdx.x >> 5;
  const int lo = lane & 15, hi = lane >> 4;
  const int m0   = blockIdx.x * 32;
  const int col0 = blockIdx.y * 256 + wave * 64;

  v8f acc[2][4];
#pragma unroll
  for (int nt = 0; nt < 4; ++nt) {
    float bias = b_proj[col0 + nt * 16 + lo];
#pragma unroll
    for (int mt = 0; mt < 2; ++mt)
#pragma unroll
      for (int i = 0; i < 8; ++i) acc[mt][nt][i] = bias;
  }

  const bf16* ar0 = aout + (size_t)(m0 + lo) * C_;
  const bf16* ar1 = aout + (size_t)(m0 + 16 + lo) * C_;
  const bf16* br[4];
#pragma unroll
  for (int nt = 0; nt < 4; ++nt) br[nt] = wbf + (size_t)(col0 + nt * 16 + lo) * C_;

  uint4 ca[4], cb[8];
  ca[0] = *(const uint4*)(ar0 + hi * 8);
  ca[1] = *(const uint4*)(ar0 + 16 + hi * 8);
  ca[2] = *(const uint4*)(ar1 + hi * 8);
  ca[3] = *(const uint4*)(ar1 + 16 + hi * 8);
#pragma unroll
  for (int nt = 0; nt < 4; ++nt) {
    cb[2*nt]   = *(const uint4*)(br[nt] + hi * 8);
    cb[2*nt+1] = *(const uint4*)(br[nt] + 16 + hi * 8);
  }

  for (int k0 = 0; k0 < C_; k0 += 32) {
    const int kn = (k0 + 32 < C_) ? (k0 + 32) : 0;
    uint4 na[4], nb[8];
    na[0] = *(const uint4*)(ar0 + kn + hi * 8);
    na[1] = *(const uint4*)(ar0 + kn + 16 + hi * 8);
    na[2] = *(const uint4*)(ar1 + kn + hi * 8);
    na[3] = *(const uint4*)(ar1 + kn + 16 + hi * 8);
#pragma unroll
    for (int nt = 0; nt < 4; ++nt) {
      nb[2*nt]   = *(const uint4*)(br[nt] + kn + hi * 8);
      nb[2*nt+1] = *(const uint4*)(br[nt] + kn + 16 + hi * 8);
    }

    v16bf a0 = pack2(ca[0], ca[1]);
    v16bf a1 = pack2(ca[2], ca[3]);
#pragma unroll
    for (int nt = 0; nt < 4; ++nt) {
      v16bf bb = pack2(cb[2*nt], cb[2*nt+1]);
      acc[0][nt] = WMMA_BF16(a0, bb, acc[0][nt]);
      acc[1][nt] = WMMA_BF16(a1, bb, acc[1][nt]);
    }
#pragma unroll
    for (int i = 0; i < 4; ++i) ca[i] = na[i];
#pragma unroll
    for (int i = 0; i < 8; ++i) cb[i] = nb[i];
  }

#pragma unroll
  for (int nt = 0; nt < 4; ++nt)
#pragma unroll
    for (int mt = 0; mt < 2; ++mt)
#pragma unroll
      for (int v = 0; v < 8; ++v)
        out[(size_t)(m0 + mt * 16 + v + 8 * hi) * C_ + col0 + nt * 16 + lo] = acc[mt][nt][v];
}

extern "C" void kernel_launch(void* const* d_in, const int* in_sizes, int n_in,
                              void* d_out, int out_size, void* d_ws, size_t ws_size,
                              hipStream_t stream) {
  (void)in_sizes; (void)n_in; (void)out_size; (void)ws_size;
  const float* x      = (const float*)d_in[0];
  const float* w_qkv  = (const float*)d_in[1];
  const float* q_bias = (const float*)d_in[2];
  const float* v_bias = (const float*)d_in[3];
  const float* w_proj = (const float*)d_in[4];
  const float* b_proj = (const float*)d_in[5];
  float* out = (float*)d_out;

  const size_t per = (size_t)B_ * H_ * N_ * D_;      // 6,291,456
  bf16* qws    = (bf16*)d_ws;
  bf16* kws    = qws  + per;
  bf16* vtws   = kws  + per;
  bf16* aoutws = vtws + per;                         // [B, N, H*D]
  bf16* xbf    = aoutws + per;                       // B*N*C == per
  bf16* wqkvbf = xbf + per;                          // 2304*768
  bf16* wprojbf = wqkvbf + (size_t)QKV_ * C_;        // 768*768

  const int nx = B_ * N_ * C_ / 8;       // 786432
  const int nw1 = QKV_ * C_ / 8;         // 221184
  const int nw2 = C_ * C_ / 8;           // 73728
  cvt_bf16_kernel<<<(nx  + 255) / 256, 256, 0, stream>>>(x,      xbf,     nx);
  cvt_bf16_kernel<<<(nw1 + 255) / 256, 256, 0, stream>>>(w_qkv,  wqkvbf,  nw1);
  cvt_bf16_kernel<<<(nw2 + 255) / 256, 256, 0, stream>>>(w_proj, wprojbf, nw2);

  qkv_gemm_kernel<<<dim3((B_ * N_) / 32, QKV_ / 256), 128, 0, stream>>>(
      xbf, wqkvbf, q_bias, v_bias, qws, kws, vtws);
  flash_attn_kernel<<<dim3((B_ * H_ * (N_ / 16)) / 4), 128, 0, stream>>>(
      qws, kws, vtws, aoutws);
  proj_gemm_kernel<<<dim3((B_ * N_) / 32, C_ / 256), 128, 0, stream>>>(
      aoutws, wprojbf, b_proj, out);
}
